// LlamaAttention_47794396070017
// MI455X (gfx1250) — matrix operand order repaired and checked by reading, not compiled, not run on hardware
//
#include <hip/hip_runtime.h>
#include <hip/hip_bf16.h>
#include <stdint.h>

typedef __bf16 bf16_t;
typedef __attribute__((ext_vector_type(16))) __bf16 v16bf;
typedef __attribute__((ext_vector_type(8)))  float  v8f;
typedef int v4i_vs __attribute__((vector_size(16)));   // matches builtin param type

union V16 { v16bf v; uint4 q[2]; bf16_t e[16]; };
union V8  { v8f  v; float f[8]; };
union V32 { uint4 q[4]; bf16_t e[32]; };

#define BATCH  2
#define SEQ    2048
#define DMODEL 2048
#define NH     16
#define NKV    4
#define HDIM   128

// ---------------------------------------------------------------- async copy
// CDNA5 async global->LDS DMA (ASYNCcnt). Builtin when available, else ISA asm.
__device__ __forceinline__ void async_b128(void* lds, const void* g) {
#if __has_builtin(__builtin_amdgcn_global_load_async_to_lds_b128)
  __builtin_amdgcn_global_load_async_to_lds_b128(
      (__attribute__((address_space(1))) v4i_vs*)(g),
      (__attribute__((address_space(3))) v4i_vs*)(lds), 0, 0);
#else
  uint32_t loff = (uint32_t)(uintptr_t)(__attribute__((address_space(3))) void*)lds;
  asm volatile("global_load_async_to_lds_b128 %0, %1, off"
               :: "v"(loff), "v"(g) : "memory");
#endif
}

__device__ __forceinline__ void wait_async0() {
#if __has_builtin(__builtin_amdgcn_s_wait_asynccnt)
  __builtin_amdgcn_s_wait_asynccnt(0);
#else
  asm volatile("s_wait_asynccnt 0x0" ::: "memory");
#endif
}

// ---------------------------------------------------------------- cast
__global__ void cast_f32_to_bf16(const float* __restrict__ in,
                                 bf16_t* __restrict__ out, int n) {
  int i = blockIdx.x * blockDim.x + threadIdx.x;
  int stride = gridDim.x * blockDim.x;
  for (; i < n; i += stride) out[i] = (bf16_t)in[i];
}

// ---------------------------------------------------------------- GEMM
// C[M][N] = A[M][K] (bf16, row-major) x W[N][K]^T (bf16, row-major)
// 128x128 block, 8 waves (2x4), 64x32 wave tile. Double-buffered LDS with
// async-to-LDS staging: wait_asynccnt -> barrier -> issue next -> compute.
__global__ __launch_bounds__(256) void gemm_bf16_wmma(
    const bf16_t* __restrict__ A, const bf16_t* __restrict__ W,
    float* __restrict__ Cf, bf16_t* __restrict__ Cb,
    int M, int N, int K) {
  __shared__ bf16_t As[2][128][32];      // 16KB
  __shared__ bf16_t Bs[2][128][32];      // 16KB

  const int tid  = threadIdx.x;
  const int lane = tid & 31;
  const int wave = tid >> 5;
  const int wm   = wave & 1;             // 2 waves along M
  const int wn   = wave >> 1;            // 4 waves along N
  const int half = lane >> 4;
  const int l16  = lane & 15;
  const int bm = blockIdx.y * 128;
  const int bn = blockIdx.x * 128;

  auto stage = [&](int buf, int k0) {
#pragma unroll
    for (int i = 0; i < 2; ++i) {
      int ch  = tid + i * 256;           // 0..511 chunk id
      int row = ch >> 2;
      int col = (ch & 3) * 8;
      async_b128(&As[buf][row][col], &A[(size_t)(bm + row) * K + k0 + col]);
      async_b128(&Bs[buf][row][col], &W[(size_t)(bn + row) * K + k0 + col]);
    }
  };

  V8 acc[4][2];
#pragma unroll
  for (int t = 0; t < 4; ++t)
#pragma unroll
    for (int c = 0; c < 2; ++c)
#pragma unroll
      for (int r = 0; r < 8; ++r) acc[t][c].f[r] = 0.f;

  const int nk = K >> 5;
  stage(0, 0);
  for (int kk = 0; kk < nk; ++kk) {
    const int buf = kk & 1;
    wait_async0();                       // my DMA for this buffer landed
    __syncthreads();                     // everyone's landed; prev buffer free
    if (kk + 1 < nk) stage(buf ^ 1, (kk + 1) << 5);

    V16 bfrag[2];
#pragma unroll
    for (int c = 0; c < 2; ++c) {        // B frag: K {0..7,16..23} / {8..15,24..31}
      const bf16_t* bp = &Bs[buf][wn * 32 + c * 16 + l16][half * 8];
      bfrag[c].q[0] = *(const uint4*)bp;
      bfrag[c].q[1] = *(const uint4*)(bp + 16);
    }
#pragma unroll
    for (int t = 0; t < 4; ++t) {        // A frag: K {0..7,16..23} / {8..15,24..31}
      V16 afrag;
      const bf16_t* ap = &As[buf][wm * 64 + t * 16 + l16][half * 8];
      afrag.q[0] = *(const uint4*)ap;
      afrag.q[1] = *(const uint4*)(ap + 16);
#pragma unroll
      for (int c = 0; c < 2; ++c)
        acc[t][c].v = __builtin_amdgcn_wmma_f32_16x16x32_bf16(
            false, afrag.v, false, bfrag[c].v, (short)0, acc[t][c].v, false, false);
    }
  }

#pragma unroll
  for (int t = 0; t < 4; ++t)
#pragma unroll
    for (int c = 0; c < 2; ++c)
#pragma unroll
      for (int r = 0; r < 8; ++r) {
        int gm = bm + wm * 64 + t * 16 + r + half * 8;
        int gn = bn + wn * 32 + c * 16 + l16;
        if (Cf) Cf[(size_t)gm * N + gn] = acc[t][c].f[r];
        else    Cb[(size_t)gm * N + gn] = (bf16_t)acc[t][c].f[r];
      }
}

// ---------------------------------------------------------------- RoPE
__global__ __launch_bounds__(64) void rope_kernel(
    bf16_t* __restrict__ Q, bf16_t* __restrict__ K,
    const float* __restrict__ cosT, const float* __restrict__ sinT) {
  const int bs = blockIdx.x;
  const int d  = threadIdx.x;            // 0..63
  const float c1 = cosT[(size_t)bs * HDIM + d];
  const float s1 = sinT[(size_t)bs * HDIM + d];
  const float c2 = cosT[(size_t)bs * HDIM + d + 64];
  const float s2 = sinT[(size_t)bs * HDIM + d + 64];

  bf16_t* qrow = Q + (size_t)bs * (NH * HDIM);
#pragma unroll
  for (int h = 0; h < NH; ++h) {
    bf16_t* p = qrow + h * HDIM;
    float x1 = (float)p[d], x2 = (float)p[d + 64];
    p[d]      = (bf16_t)(x1 * c1 - x2 * s1);
    p[d + 64] = (bf16_t)(x2 * c2 + x1 * s2);
  }
  bf16_t* krow = K + (size_t)bs * (NKV * HDIM);
#pragma unroll
  for (int h = 0; h < NKV; ++h) {
    bf16_t* p = krow + h * HDIM;
    float x1 = (float)p[d], x2 = (float)p[d + 64];
    p[d]      = (bf16_t)(x1 * c1 - x2 * s1);
    p[d + 64] = (bf16_t)(x2 * c2 + x1 * s2);
  }
}

// ---------------------------------------------------------------- flash attn
// grid: (SEQ/64, NH, B). 4 waves; wave w owns q rows qblk*64 + w*16 .. +15.
// Double-buffered K (async DMA) / V (transposed via VGPRs) tiles.
__global__ __launch_bounds__(128) void flash_attn_wmma(
    const bf16_t* __restrict__ Q, const bf16_t* __restrict__ Kc,
    const bf16_t* __restrict__ Vc, bf16_t* __restrict__ O) {
  __shared__ bf16_t Ks[2][32][HDIM];     // 16KB [buf][key][hd]
  __shared__ bf16_t Vt[2][HDIM][32];     // 16KB [buf][hd][key] (transposed)
  __shared__ bf16_t Ps[4][16][32];       //  4KB per-wave P tile

  const int qblk = blockIdx.x, h = blockIdx.y, b = blockIdx.z;
  const int kvh  = h >> 2;               // GQA: 4 q-heads per kv-head
  const int tid  = threadIdx.x, lane = tid & 31, wave = tid >> 5;
  const int half = lane >> 4, l16 = lane & 15;
  const int qrow0 = qblk * 64 + wave * 16;

  auto stage = [&](int buf, int kb) {
    int row = tid >> 2, col = (tid & 3) * 32;
    const bf16_t* kp = Kc + ((size_t)(b * SEQ) + kb * 32 + row) * (NKV * HDIM)
                          + kvh * HDIM + col;
    async_b128(&Ks[buf][row][col],      kp);
    async_b128(&Ks[buf][row][col + 8],  kp + 8);
    async_b128(&Ks[buf][row][col + 16], kp + 16);
    async_b128(&Ks[buf][row][col + 24], kp + 24);

    const bf16_t* vp = Vc + ((size_t)(b * SEQ) + kb * 32 + row) * (NKV * HDIM)
                          + kvh * HDIM + col;
    V32 vv;
    vv.q[0] = *(const uint4*)(vp);
    vv.q[1] = *(const uint4*)(vp + 8);
    vv.q[2] = *(const uint4*)(vp + 16);
    vv.q[3] = *(const uint4*)(vp + 24);
#pragma unroll
    for (int i = 0; i < 32; ++i) Vt[buf][col + i][row] = vv.e[i];
  };

  // Q fragments for this wave's 16 rows (4 K-steps over hd=128)
  V16 qf[4];
  {
    const bf16_t* qp = Q + ((size_t)(b * SEQ) + qrow0 + l16) * (NH * HDIM) + h * HDIM;
#pragma unroll
    for (int ks = 0; ks < 4; ++ks) {
      qf[ks].q[0] = *(const uint4*)(qp + ks * 32 + half * 8);
      qf[ks].q[1] = *(const uint4*)(qp + ks * 32 + 16 + half * 8);
    }
  }

  V8 o[8];
  float mrow[8], lrow[8];
#pragma unroll
  for (int t = 0; t < 8; ++t)
#pragma unroll
    for (int r = 0; r < 8; ++r) o[t].f[r] = 0.f;
#pragma unroll
  for (int r = 0; r < 8; ++r) { mrow[r] = -1e30f; lrow[r] = 0.f; }

  const float scale = 0.08838834764831845f;  // 1/sqrt(128)
  const int nkb = 2 * qblk + 2;              // causal: keys 0 .. qblk*64+63

  stage(0, 0);
  for (int kb = 0; kb < nkb; ++kb) {
    const int buf = kb & 1;
    wait_async0();
    __syncthreads();                     // K DMA + V ds_stores published
    if (kb + 1 < nkb) stage(buf ^ 1, kb + 1);

    // S = Q K^T  (two 16x16 column tiles over this 32-key block)
    V8 s0, s1;
#pragma unroll
    for (int r = 0; r < 8; ++r) { s0.f[r] = 0.f; s1.f[r] = 0.f; }
#pragma unroll
    for (int ks = 0; ks < 4; ++ks) {
      V16 kf0, kf1;
      const bf16_t* bp0 = &Ks[buf][l16][ks * 32 + half * 8];
      kf0.q[0] = *(const uint4*)bp0;
      kf0.q[1] = *(const uint4*)(bp0 + 16);
      const bf16_t* bp1 = &Ks[buf][16 + l16][ks * 32 + half * 8];
      kf1.q[0] = *(const uint4*)bp1;
      kf1.q[1] = *(const uint4*)(bp1 + 16);
      s0.v = __builtin_amdgcn_wmma_f32_16x16x32_bf16(false, qf[ks].v, false, kf0.v,
                                                     (short)0, s0.v, false, false);
      s1.v = __builtin_amdgcn_wmma_f32_16x16x32_bf16(false, qf[ks].v, false, kf1.v,
                                                     (short)0, s1.v, false, false);
    }

    // online softmax (row = qrow0 + r + half*8 ; a row lives in one 16-lane half)
    float p0[8], p1[8];
#pragma unroll
    for (int r = 0; r < 8; ++r) {
      const int row  = qrow0 + r + half * 8;
      const int col0 = kb * 32 + l16;
      const int col1 = col0 + 16;
      float a = s0.f[r] * scale; if (col0 > row) a = -1e30f;
      float c = s1.f[r] * scale; if (col1 > row) c = -1e30f;
      float rmax = fmaxf(a, c);
#pragma unroll
      for (int off = 8; off >= 1; off >>= 1)
        rmax = fmaxf(rmax, __shfl_xor(rmax, off));
      float mnew  = fmaxf(mrow[r], rmax);
      float alpha = __expf(mrow[r] - mnew);
      mrow[r] = mnew;
      p0[r] = __expf(a - mnew);
      p1[r] = __expf(c - mnew);
      float rsum = p0[r] + p1[r];
#pragma unroll
      for (int off = 8; off >= 1; off >>= 1)
        rsum += __shfl_xor(rsum, off);
      lrow[r] = lrow[r] * alpha + rsum;
#pragma unroll
      for (int t = 0; t < 8; ++t) o[t].f[r] *= alpha;
    }

    // C-layout -> A-layout via wave-private LDS tile
#pragma unroll
    for (int r = 0; r < 8; ++r) {
      Ps[wave][r + half * 8][l16]      = (bf16_t)p0[r];
      Ps[wave][r + half * 8][16 + l16] = (bf16_t)p1[r];
    }
    V16 pf;
    {
      const bf16_t* pp = &Ps[wave][l16][half * 8];
      pf.q[0] = *(const uint4*)pp;
      pf.q[1] = *(const uint4*)(pp + 16);
    }

    // O += P (16x32) * V (32x128) : 8 hd tiles
#pragma unroll
    for (int t = 0; t < 8; ++t) {
      V16 vf;
      const bf16_t* vp2 = &Vt[buf][t * 16 + l16][half * 8];
      vf.q[0] = *(const uint4*)vp2;
      vf.q[1] = *(const uint4*)(vp2 + 16);
      o[t].v = __builtin_amdgcn_wmma_f32_16x16x32_bf16(false, pf.v, false, vf.v,
                                                       (short)0, o[t].v, false, false);
    }
  }

  // normalize + store: out layout [b][s][h*128+hd]
#pragma unroll
  for (int t = 0; t < 8; ++t)
#pragma unroll
    for (int r = 0; r < 8; ++r) {
      int gr = qrow0 + r + half * 8;
      O[((size_t)(b * SEQ) + gr) * (NH * HDIM) + h * HDIM + t * 16 + l16] =
          (bf16_t)(o[t].f[r] / lrow[r]);
    }
}

// ---------------------------------------------------------------- launch
extern "C" void kernel_launch(void* const* d_in, const int* in_sizes, int n_in,
                              void* d_out, int out_size, void* d_ws, size_t ws_size,
                              hipStream_t stream) {
  const float* hidden = (const float*)d_in[0];
  const float* cosT   = (const float*)d_in[1];
  const float* sinT   = (const float*)d_in[2];
  const float* Wq     = (const float*)d_in[3];
  const float* Wk     = (const float*)d_in[4];
  const float* Wv     = (const float*)d_in[5];
  const float* Wo     = (const float*)d_in[6];
  float* out = (float*)d_out;

  const int M = BATCH * SEQ;             // 4096 rows, batch-flattened
  char* ws = (char*)d_ws;
  size_t off = 0;
  auto alloc = [&](size_t bytes) {
    char* p = ws + off;
    off += (bytes + 255) & ~(size_t)255;
    return p;
  };
  bf16_t* hb  = (bf16_t*)alloc((size_t)M * DMODEL * 2);        // 16 MB
  bf16_t* wqb = (bf16_t*)alloc((size_t)(NH * HDIM) * DMODEL * 2);
  bf16_t* wkb = (bf16_t*)alloc((size_t)(NKV * HDIM) * DMODEL * 2);
  bf16_t* wvb = (bf16_t*)alloc((size_t)(NKV * HDIM) * DMODEL * 2);
  bf16_t* wob = (bf16_t*)alloc((size_t)DMODEL * DMODEL * 2);
  bf16_t* qb  = (bf16_t*)alloc((size_t)M * NH * HDIM * 2);
  bf16_t* kb  = (bf16_t*)alloc((size_t)M * NKV * HDIM * 2);
  bf16_t* vb  = (bf16_t*)alloc((size_t)M * NKV * HDIM * 2);
  bf16_t* aob = (bf16_t*)alloc((size_t)M * NH * HDIM * 2);
  (void)ws_size; (void)in_sizes; (void)n_in; (void)out_size;

  // 1) fp32 -> bf16 casts (L2-resident)
  cast_f32_to_bf16<<<2048, 256, 0, stream>>>(hidden, hb, M * DMODEL);
  cast_f32_to_bf16<<<1024, 256, 0, stream>>>(Wq, wqb, NH * HDIM * DMODEL);
  cast_f32_to_bf16<<<512,  256, 0, stream>>>(Wk, wkb, NKV * HDIM * DMODEL);
  cast_f32_to_bf16<<<512,  256, 0, stream>>>(Wv, wvb, NKV * HDIM * DMODEL);
  cast_f32_to_bf16<<<1024, 256, 0, stream>>>(Wo, wob, DMODEL * DMODEL);

  // 2) QKV projections (WMMA bf16)
  dim3 gq(DMODEL / 128, M / 128);                        // (16, 32)
  dim3 gkv((NKV * HDIM) / 128, M / 128);                 // (4, 32)
  gemm_bf16_wmma<<<gq,  256, 0, stream>>>(hb, wqb, nullptr, qb, M, NH * HDIM, DMODEL);
  gemm_bf16_wmma<<<gkv, 256, 0, stream>>>(hb, wkb, nullptr, kb, M, NKV * HDIM, DMODEL);
  gemm_bf16_wmma<<<gkv, 256, 0, stream>>>(hb, wvb, nullptr, vb, M, NKV * HDIM, DMODEL);

  // 3) RoPE on Q and K
  rope_kernel<<<M, 64, 0, stream>>>(qb, kb, cosT, sinT);

  // 4) causal flash attention (GQA), WMMA for QK^T and PV
  dim3 ga(SEQ / 64, NH, BATCH);
  flash_attn_wmma<<<ga, 128, 0, stream>>>(qb, kb, vb, aob);

  // 5) output projection -> fp32 d_out
  gemm_bf16_wmma<<<gq, 256, 0, stream>>>(aob, wob, out, nullptr, M, DMODEL, DMODEL);
}
